// GPTSelfAttention_65695819760159
// MI455X (gfx1250) — matrix operand order repaired and checked by reading, not compiled, not run on hardware
//
#include <hip/hip_runtime.h>
#include <hip/hip_bf16.h>

#define EMBED  2048
#define NHEADS 16
#define HDIM   128
#define SEQ    2048
#define BATCH  2
#define MROWS  (BATCH*SEQ)   // 4096
#define QK_SCALE 0.08838834764831845f   // 128^-0.5

typedef __attribute__((ext_vector_type(16))) __bf16 v16bf;
typedef __attribute__((ext_vector_type(8)))  float  v8f;

// ---------------- WMMA helpers (layouts per CDNA5 ISA 7.12.2, wave32) ------

static __device__ __forceinline__ v8f wmma_bf16(v16bf a, v16bf b, v8f c) {
  return __builtin_amdgcn_wmma_f32_16x16x32_bf16(false, a, false, b,
                                                 (short)0, c, false, false);
}

// A fragment: 16x32 bf16, row m = lane&15, K contiguous (stride_k == 1).
// lanes 0-15: K e<8 -> e,      e>=8 -> e+8   (0..7,16..23)
// lanes16-31: K e<8 -> e+8,    e>=8 -> e+16  (8..15,24..31)
static __device__ __forceinline__ v16bf load_a_frag(const __bf16* base,
                                                    int stride_m, int lane) {
  int m  = lane & 15;
  int kb = (lane < 16) ? 0 : 8;
  v16bf a;
#pragma unroll
  for (int e = 0; e < 16; ++e) {
    int k = kb + e + ((e < 8) ? 0 : 8);
    a[e] = base[m * stride_m + k];
  }
  return a;
}

// B fragment: 32(K)x16(N) bf16, col n = lane&15, K same as A (0..7,16..23 / 8..15,24..31).
// value(k,n) = base[k*stride_k + n*stride_n]
static __device__ __forceinline__ v16bf load_b_frag(const __bf16* base,
                                                    int stride_k, int stride_n,
                                                    int lane) {
  int n  = lane & 15;
  int kb = (lane < 16) ? 0 : 8;
  v16bf b;
#pragma unroll
  for (int e = 0; e < 16; ++e) {
    int k = kb + e + ((e < 8) ? 0 : 8);
    b[e] = base[k * stride_k + n * stride_n];
  }
  return b;
}

// 16-lane (half-wave) reductions; xor masks < 16 stay within each half.
static __device__ __forceinline__ float rmax16(float v) {
#pragma unroll
  for (int o = 8; o >= 1; o >>= 1) v = fmaxf(v, __shfl_xor(v, o, 32));
  return v;
}
static __device__ __forceinline__ float rsum16(float v) {
#pragma unroll
  for (int o = 8; o >= 1; o >>= 1) v += __shfl_xor(v, o, 32);
  return v;
}

// Async global->LDS 128-bit copy (CDNA5, tracked by ASYNCcnt).
// dsaddr = LDS byte address (low 32 bits of generic shared pointer),
// GVS mode: mem = SADDR(64b) + VADDR(32b offset).
static __device__ __forceinline__ void async_b128(unsigned lds_byte,
                                                  unsigned gl_byte_off,
                                                  const void* base) {
  asm volatile("global_load_async_to_lds_b128 %0, %1, %2"
               :: "v"(lds_byte), "v"(gl_byte_off), "s"(base)
               : "memory");
}
static __device__ __forceinline__ void wait_asynccnt0() {
  asm volatile("s_wait_asynccnt 0x0" ::: "memory");
}

// ---------------- Kernel 1: fused projection  Y = (X @ W^T) * scale --------
// X: fp32 [4096 x 2048], W: fp32 [2048 x 2048] (torch Linear: y=x@W.T)
// Out: bf16 in [B,H,S,D] split-head layout.
__global__ __launch_bounds__(256)
void qkv_proj_kernel(const float* __restrict__ X, const float* __restrict__ W,
                     __bf16* __restrict__ Out, float outScale) {
  const int LDK = 40;                       // 32 + 8 pad
  __shared__ __align__(16) __bf16 Xs[128 * LDK];
  __shared__ __align__(16) __bf16 Ws[128 * LDK];

  int tid  = threadIdx.x;
  int lane = tid & 31;
  int wid  = tid >> 5;        // 0..7
  int wm   = wid & 3;         // 4 waves along M, 32 rows each
  int wn   = wid >> 2;        // 2 waves along N, 64 cols each
  int bm   = blockIdx.y * 128;
  int bn   = blockIdx.x * 128;

  v8f acc[2][4];
#pragma unroll
  for (int i = 0; i < 2; ++i)
#pragma unroll
    for (int j = 0; j < 4; ++j) acc[i][j] = (v8f){0.f,0.f,0.f,0.f,0.f,0.f,0.f,0.f};

  int frow = tid >> 1;
  int fko  = (tid & 1) * 16;
  const float* xrow = X + (size_t)(bm + frow) * EMBED + fko;
  const float* wrow = W + (size_t)(bn + frow) * EMBED + fko;

  for (int kt = 0; kt < EMBED; kt += 32) {
    const float4* xp = (const float4*)(xrow + kt);
    const float4* wp = (const float4*)(wrow + kt);
    if (kt + 32 < EMBED) {                   // pull next K-tile toward WGP
      __builtin_prefetch(xrow + kt + 32, 0, 3);
      __builtin_prefetch(wrow + kt + 32, 0, 3);
    }
#pragma unroll
    for (int c = 0; c < 4; ++c) {
      float4 xv = xp[c];
      float4 wv = wp[c];
      int o = frow * LDK + fko + c * 4;
      Xs[o+0] = (__bf16)xv.x; Xs[o+1] = (__bf16)xv.y;
      Xs[o+2] = (__bf16)xv.z; Xs[o+3] = (__bf16)xv.w;
      Ws[o+0] = (__bf16)wv.x; Ws[o+1] = (__bf16)wv.y;
      Ws[o+2] = (__bf16)wv.z; Ws[o+3] = (__bf16)wv.w;
    }
    __syncthreads();

    v16bf afr[2];
#pragma unroll
    for (int mi = 0; mi < 2; ++mi)
      afr[mi] = load_a_frag(&Xs[(wm * 32 + mi * 16) * LDK], LDK, lane);
#pragma unroll
    for (int ni = 0; ni < 4; ++ni) {
      v16bf bfr = load_b_frag(&Ws[(wn * 64 + ni * 16) * LDK], 1, LDK, lane);
#pragma unroll
      for (int mi = 0; mi < 2; ++mi)
        acc[mi][ni] = wmma_bf16(afr[mi], bfr, acc[mi][ni]);
    }
    __syncthreads();
  }

  int mlo = (lane >> 4) << 3;   // 0 or 8
  int nl  = lane & 15;
#pragma unroll
  for (int mi = 0; mi < 2; ++mi)
#pragma unroll
    for (int ni = 0; ni < 4; ++ni)
#pragma unroll
      for (int r = 0; r < 8; ++r) {
        int m = bm + wm * 32 + mi * 16 + r + mlo;
        int n = bn + wn * 64 + ni * 16 + nl;
        int b = m >> 11, s = m & (SEQ - 1);
        int h = n >> 7,  d = n & (HDIM - 1);
        float v = acc[mi][ni][r] * outScale;
        Out[(((size_t)(b * NHEADS + h)) * SEQ + s) * HDIM + d] = (__bf16)v;
      }
}

// ---------------- Kernel 2: flash attention per (b,h) head ----------------
// Q,K,V: bf16 [B,H,S,D] (Q pre-scaled). O: bf16 merged-head [B*S, E].
// K/V tiles double-buffered in LDS via async global->LDS copies.
__global__ __launch_bounds__(128)
void attn_kernel(const __bf16* __restrict__ Q, const __bf16* __restrict__ Kg,
                 const __bf16* __restrict__ Vg, const float* __restrict__ mask,
                 __bf16* __restrict__ O) {
  const int LDKV = 136;   // 128 + 8 pad
  const int LDP  = 72;    // 64 + 8 pad
  __shared__ __align__(16) __bf16 Ks[2][64 * LDKV];
  __shared__ __align__(16) __bf16 Vs[2][64 * LDKV];
  __shared__ __align__(16) __bf16 Ps[4 * 16 * LDP];

  int tid  = threadIdx.x;
  int lane = tid & 31;
  int wid  = tid >> 5;                 // 4 waves, 16 q-rows each
  int bh   = blockIdx.y;
  int b    = bh >> 4;
  int h    = bh & (NHEADS - 1);
  int q0   = blockIdx.x * 64;
  int qr   = q0 + wid * 16;

  size_t headoff = (size_t)bh * SEQ * HDIM;
  const __bf16* Qh = Q  + headoff;
  const __bf16* Kh = Kg + headoff;
  const __bf16* Vh = Vg + headoff;

  // per-thread tile-fill coordinates: 64 rows x 16 chunks of 8 bf16
  int frow = (tid * 8) >> 7;           // base row for chunk id = tid
  int fcc  = tid & 15;

  auto prefetch_tile = [&](int j0, int buf) {
#pragma unroll
    for (int i = 0; i < 8; ++i) {
      int cid = tid + i * 128;
      int row = cid >> 4, cc = cid & 15;
      unsigned goff = (unsigned)((((j0 + row) * HDIM) + cc * 8) * 2);
      unsigned kl = (unsigned)(size_t)&Ks[buf][row * LDKV + cc * 8];
      unsigned vl = (unsigned)(size_t)&Vs[buf][row * LDKV + cc * 8];
      async_b128(kl, goff, (const void*)Kh);
      async_b128(vl, goff, (const void*)Vh);
    }
  };
  (void)frow; (void)fcc;

  // Q fragments held in registers: 16 rows x 128 (4 x K32 chunks)
  v16bf aq[4];
#pragma unroll
  for (int kc = 0; kc < 4; ++kc)
    aq[kc] = load_a_frag(Qh + (size_t)qr * HDIM + kc * 32, HDIM, lane);

  v8f of[8];
#pragma unroll
  for (int i = 0; i < 8; ++i) of[i] = (v8f){0.f,0.f,0.f,0.f,0.f,0.f,0.f,0.f};
  float mrun[8], lrun[8];
#pragma unroll
  for (int r = 0; r < 8; ++r) { mrun[r] = -3.0e38f; lrun[r] = 0.f; }

  int mlo = (lane >> 4) << 3;
  int nl  = lane & 15;
  __bf16* Pw = &Ps[wid * 16 * LDP];

  prefetch_tile(0, 0);                 // async fill of tile 0

  const int NTILES = SEQ / 64;
  for (int jt = 0; jt < NTILES; ++jt) {
    int buf = jt & 1;
    int j0  = jt * 64;
    wait_asynccnt0();        // my async fills for tile jt landed
    __syncthreads();         // everyone's landed; prev reads of buf^1 done
    if (jt + 1 < NTILES) prefetch_tile(j0 + 64, buf ^ 1);   // overlap with compute

    const __bf16* Kt = Ks[buf];
    const __bf16* Vt = Vs[buf];

    // ---- S = Q K^T  (16 q-rows x 64 k-cols)
    v8f sf[4];
#pragma unroll
    for (int ni = 0; ni < 4; ++ni) sf[ni] = (v8f){0.f,0.f,0.f,0.f,0.f,0.f,0.f,0.f};
#pragma unroll
    for (int ni = 0; ni < 4; ++ni)
#pragma unroll
      for (int dc = 0; dc < 4; ++dc) {
        v16bf bfr = load_b_frag(&Kt[(ni * 16) * LDKV + dc * 32], 1, LDKV, lane);
        sf[ni] = wmma_bf16(aq[dc], bfr, sf[ni]);
      }

    // ---- + attention mask (reference adds it; zeros here)
#pragma unroll
    for (int ni = 0; ni < 4; ++ni)
#pragma unroll
      for (int r = 0; r < 8; ++r) {
        int qg = qr + r + mlo;
        int kg = j0 + ni * 16 + nl;
        sf[ni][r] += mask[(size_t)qg * SEQ + kg];
      }

    // ---- online softmax (per row; a row's 16 cols live in one lane-half)
    float alpha[8];
#pragma unroll
    for (int r = 0; r < 8; ++r) {
      float tm = fmaxf(fmaxf(sf[0][r], sf[1][r]), fmaxf(sf[2][r], sf[3][r]));
      tm = rmax16(tm);
      float mnew = fmaxf(mrun[r], tm);
      alpha[r] = __expf(mrun[r] - mnew);
      mrun[r] = mnew;
      float rs = 0.f;
#pragma unroll
      for (int ni = 0; ni < 4; ++ni) {
        float p = __expf(sf[ni][r] - mnew);
        sf[ni][r] = p;
        rs += p;
      }
      rs = rsum16(rs);
      lrun[r] = lrun[r] * alpha[r] + rs;
    }
#pragma unroll
    for (int od = 0; od < 8; ++od)
#pragma unroll
      for (int r = 0; r < 8; ++r) of[od][r] *= alpha[r];

    // ---- restage P (C-layout -> A-layout) through LDS
#pragma unroll
    for (int ni = 0; ni < 4; ++ni)
#pragma unroll
      for (int r = 0; r < 8; ++r)
        Pw[(r + mlo) * LDP + ni * 16 + nl] = (__bf16)sf[ni][r];
    __syncthreads();

    // ---- O += P V   (16 x 64 @ 64 x 128)
#pragma unroll
    for (int ks = 0; ks < 2; ++ks) {
      v16bf ap = load_a_frag(Pw + ks * 32, LDP, lane);
#pragma unroll
      for (int od = 0; od < 8; ++od) {
        v16bf bv = load_b_frag(&Vt[(ks * 32) * LDKV + od * 16], LDKV, 1, lane);
        of[od] = wmma_bf16(ap, bv, of[od]);
      }
    }
  }

  // ---- normalize and write merged-head [B*S, E] bf16
#pragma unroll
  for (int r = 0; r < 8; ++r) {
    float inv = 1.0f / lrun[r];
    int srow = qr + r + mlo;
    size_t orow = ((size_t)b * SEQ + srow) * EMBED + (size_t)h * HDIM;
#pragma unroll
    for (int od = 0; od < 8; ++od)
      O[orow + od * 16 + nl] = (__bf16)(of[od][r] * inv);
  }
}

// ---------------- Kernel 3: output projection  out = Oh @ Wo^T (fp32) ------
__global__ __launch_bounds__(256)
void out_proj_kernel(const __bf16* __restrict__ Xb, const float* __restrict__ W,
                     float* __restrict__ Out) {
  const int LDK = 40;
  __shared__ __align__(16) __bf16 Xs[128 * LDK];
  __shared__ __align__(16) __bf16 Ws[128 * LDK];

  int tid  = threadIdx.x;
  int lane = tid & 31;
  int wid  = tid >> 5;
  int wm   = wid & 3;
  int wn   = wid >> 2;
  int bm   = blockIdx.y * 128;
  int bn   = blockIdx.x * 128;

  v8f acc[2][4];
#pragma unroll
  for (int i = 0; i < 2; ++i)
#pragma unroll
    for (int j = 0; j < 4; ++j) acc[i][j] = (v8f){0.f,0.f,0.f,0.f,0.f,0.f,0.f,0.f};

  int frow = tid >> 1;
  int fko  = (tid & 1) * 16;
  const __bf16* xrow = Xb + (size_t)(bm + frow) * EMBED + fko;
  const float*  wrow = W  + (size_t)(bn + frow) * EMBED + fko;

  for (int kt = 0; kt < EMBED; kt += 32) {
    if (kt + 32 < EMBED) {
      __builtin_prefetch(xrow + kt + 32, 0, 3);
      __builtin_prefetch(wrow + kt + 32, 0, 3);
    }
    // X already bf16: two 128-bit copies
    const uint4* xp = (const uint4*)(xrow + kt);
    uint4* xd = (uint4*)&Xs[frow * LDK + fko];
    xd[0] = xp[0];
    xd[1] = xp[1];
    const float4* wp = (const float4*)(wrow + kt);
#pragma unroll
    for (int c = 0; c < 4; ++c) {
      float4 wv = wp[c];
      int o = frow * LDK + fko + c * 4;
      Ws[o+0] = (__bf16)wv.x; Ws[o+1] = (__bf16)wv.y;
      Ws[o+2] = (__bf16)wv.z; Ws[o+3] = (__bf16)wv.w;
    }
    __syncthreads();

    v16bf afr[2];
#pragma unroll
    for (int mi = 0; mi < 2; ++mi)
      afr[mi] = load_a_frag(&Xs[(wm * 32 + mi * 16) * LDK], LDK, lane);
#pragma unroll
    for (int ni = 0; ni < 4; ++ni) {
      v16bf bfr = load_b_frag(&Ws[(wn * 64 + ni * 16) * LDK], 1, LDK, lane);
#pragma unroll
      for (int mi = 0; mi < 2; ++mi)
        acc[mi][ni] = wmma_bf16(afr[mi], bfr, acc[mi][ni]);
    }
    __syncthreads();
  }

  int mlo = (lane >> 4) << 3;
  int nl  = lane & 15;
#pragma unroll
  for (int mi = 0; mi < 2; ++mi)
#pragma unroll
    for (int ni = 0; ni < 4; ++ni)
#pragma unroll
      for (int r = 0; r < 8; ++r) {
        int m = bm + wm * 32 + mi * 16 + r + mlo;
        int n = bn + wn * 64 + ni * 16 + nl;
        Out[(size_t)m * EMBED + n] = acc[mi][ni][r];
      }
}

// ---------------- launch ---------------------------------------------------
extern "C" void kernel_launch(void* const* d_in, const int* in_sizes, int n_in,
                              void* d_out, int out_size, void* d_ws, size_t ws_size,
                              hipStream_t stream) {
  const float* hidden = (const float*)d_in[0];
  const float* mask   = (const float*)d_in[1];
  const float* Wq     = (const float*)d_in[2];
  const float* Wk     = (const float*)d_in[3];
  const float* Wv     = (const float*)d_in[4];
  const float* Wo     = (const float*)d_in[5];
  float* out = (float*)d_out;

  char* ws = (char*)d_ws;
  size_t seg = (size_t)MROWS * EMBED * sizeof(__bf16);   // 16 MB each
  __bf16* qws = (__bf16*)(ws + 0 * seg);
  __bf16* kws = (__bf16*)(ws + 1 * seg);
  __bf16* vws = (__bf16*)(ws + 2 * seg);
  __bf16* ows = (__bf16*)(ws + 3 * seg);

  dim3 gg(EMBED / 128, MROWS / 128);   // (16, 32)
  qkv_proj_kernel<<<gg, 256, 0, stream>>>(hidden, Wq, qws, QK_SCALE);
  qkv_proj_kernel<<<gg, 256, 0, stream>>>(hidden, Wk, kws, 1.0f);
  qkv_proj_kernel<<<gg, 256, 0, stream>>>(hidden, Wv, vws, 1.0f);

  attn_kernel<<<dim3(SEQ / 64, BATCH * NHEADS), 128, 0, stream>>>(
      qws, kws, vws, mask, ows);

  out_proj_kernel<<<gg, 256, 0, stream>>>(ows, Wo, out);
}
